// DeepseekV2Moe_20014547599362
// MI455X (gfx1250) — compile-verified, matrix operand order checked by reading, not run on hardware
//
#include <hip/hip_runtime.h>
#include <hip/hip_bf16.h>
#include <math.h>

// ---- problem constants (from reference) ----
#define NE    64      // experts
#define NK    6       // top-k
#define CAP   512     // per-expert capacity
#define HD    2048    // hidden
#define ID    768     // moe intermediate
#define TWOI  1536    // 2*ID
#define SIE   1536    // shared experts intermediate
#define NTOK  2048    // B*S
#define SCALE_F 1.0f

#define KP    34      // padded K stride (halves) for VALU-filled LDS tiles
#define KPA   40      // padded K stride (halves) for async-filled tiles (80B, 16B-aligned rows)

typedef __attribute__((ext_vector_type(16))) __bf16 v16bf;
typedef __attribute__((ext_vector_type(8)))  float  v8f;
typedef __attribute__((ext_vector_type(4)))  int    v4i;

// ---- async global->LDS copy (CDNA5 ASYNCcnt path) ----
#if defined(__has_builtin)
#if __has_builtin(__builtin_amdgcn_global_load_async_to_lds_b128)
#define ASYNC_B128_BUILTIN 1
#endif
#if __has_builtin(__builtin_amdgcn_s_wait_asynccnt)
#define WAIT_ASYNC_BUILTIN 1
#endif
#endif

__device__ __forceinline__ void async_copy_b128(const __bf16* gsrc, __bf16* ldst) {
#ifdef ASYNC_B128_BUILTIN
  __builtin_amdgcn_global_load_async_to_lds_b128(
      (__attribute__((address_space(1))) v4i*)gsrc,
      (__attribute__((address_space(3))) v4i*)ldst, 0, 0);
#else
  asm volatile("global_load_async_to_lds_b128 %0, %1, off"
               :: "v"((unsigned)(unsigned long long)ldst), "v"(gsrc)
               : "memory");
#endif
}

__device__ __forceinline__ void wait_async0() {
#ifdef WAIT_ASYNC_BUILTIN
  __builtin_amdgcn_s_wait_asynccnt(0);
#else
  asm volatile("s_wait_asynccnt 0" ::: "memory");
#endif
}

// ---- WMMA helpers (CDNA5 wave32, 16x16x32 bf16 -> f32) ----
__device__ __forceinline__ v8f wmma_bf16(v16bf a, v16bf b, v8f c) {
  return __builtin_amdgcn_wmma_f32_16x16x32_bf16(false, a, false, b, (short)0, c,
                                                 false, false);
}

// Load a 16(M/N) x 32(K) bf16 fragment from an LDS tile stored [row][k].
// ISA 7.12.2 16-bit A layout: lanes 0-15 K=0..7/16..23, lanes 16-31 K=8..15/24..31.
__device__ __forceinline__ v16bf frag_ld(const __bf16* base, int lane, int stride) {
  int r   = lane & 15;
  int khi = (lane >> 4) << 3;     // 0 or 8
  const __bf16* p = base + r * stride;
  v16bf f;
#pragma unroll
  for (int i = 0; i < 8; ++i) {
    f[i]     = p[khi + i];
    f[i + 8] = p[16 + khi + i];
  }
  return f;
}

__device__ __forceinline__ float silu_mul(float g, float u) {
  return (g / (1.0f + __expf(-g))) * u;
}

// ------------------------------------------------------------------
// 0) zero the per-expert counters
// ------------------------------------------------------------------
__global__ void k_init(int* __restrict__ counts) {
  if (threadIdx.x < NE) counts[threadIdx.x] = 0;
}

// ------------------------------------------------------------------
// 1) router: logits = x @ gate_w^T, softmax, greedy top-6, dispatch
// ------------------------------------------------------------------
__global__ __launch_bounds__(256)
void k_router(const float* __restrict__ x, const float* __restrict__ gw,
              int* __restrict__ counts, int* __restrict__ buf_tok,
              float* __restrict__ buf_w) {
  __shared__ float xs[HD];
  __shared__ float part[256];
  __shared__ float logit[NE];
  const int t = blockIdx.x, tid = threadIdx.x;
  for (int i = tid; i < HD; i += 256) xs[i] = x[(size_t)t * HD + i];
  __syncthreads();
  const int e = tid >> 2, q = tid & 3;
  const float* w  = gw + (size_t)e * HD + q * 512;
  const float* xv = xs + q * 512;
  float s = 0.f;
#pragma unroll 8
  for (int i = 0; i < 512; ++i) s = fmaf(xv[i], w[i], s);
  part[tid] = s;
  __syncthreads();
  if (q == 0) logit[e] = part[tid] + part[tid + 1] + part[tid + 2] + part[tid + 3];
  __syncthreads();
  if (tid == 0) {
    float mx = -1e30f;
    for (int i = 0; i < NE; ++i) mx = fmaxf(mx, logit[i]);
    float sum = 0.f;
    for (int i = 0; i < NE; ++i) sum += __expf(logit[i] - mx);
    const float inv = 1.0f / sum;
    for (int k = 0; k < NK; ++k) {
      int best = 0; float bv = -1e30f;
      for (int i = 0; i < NE; ++i) {
        float v = logit[i];
        if (v > bv) { bv = v; best = i; }
      }
      logit[best] = -1e30f;
      const float wgt = __expf(bv - mx) * inv * SCALE_F;
      const int pos = atomicAdd(&counts[best], 1);
      if (pos < CAP) {
        buf_tok[best * CAP + pos] = t;
        buf_w [best * CAP + pos] = wgt;
      }
    }
  }
}

// ------------------------------------------------------------------
// 2) shared experts: fused gate+up SwiGLU -> hs[T, SIE] bf16
// ------------------------------------------------------------------
__global__ __launch_bounds__(256)
void k_shared_gateup(const float* __restrict__ x, const float* __restrict__ wg,
                     const float* __restrict__ wu, __bf16* __restrict__ hs) {
  const int m0 = blockIdx.x * 128, n0 = blockIdx.y * 64;
  __shared__ __bf16 aT[128 * KP];
  __shared__ __bf16 bgT[64 * KP];
  __shared__ __bf16 buT[64 * KP];
  const int tid = threadIdx.x, lane = tid & 31, wv = tid >> 5;
  const int wm = wv & 3, wn = wv >> 2;
  const int arow = tid >> 1, aks = (tid & 1) * 16;
  const float* xrow = x + (size_t)(m0 + arow) * HD;
  const int bn = tid >> 2, bkb = (tid & 3) * 8;
  v8f cg[2][2], cu[2][2];
#pragma unroll
  for (int i = 0; i < 2; ++i)
#pragma unroll
    for (int j = 0; j < 2; ++j) { cg[i][j] = (v8f){0,0,0,0,0,0,0,0}; cu[i][j] = cg[i][j]; }

  for (int k0 = 0; k0 < HD; k0 += 32) {
#pragma unroll
    for (int j = 0; j < 16; ++j)
      aT[arow * KP + aks + j] = (__bf16)xrow[k0 + aks + j];
    const float* gp = wg + (size_t)(n0 + bn) * HD + k0 + bkb;
    const float* up = wu + (size_t)(n0 + bn) * HD + k0 + bkb;
#pragma unroll
    for (int j = 0; j < 8; ++j) {
      bgT[bn * KP + bkb + j] = (__bf16)gp[j];
      buT[bn * KP + bkb + j] = (__bf16)up[j];
    }
    if (k0 + 32 < HD) { __builtin_prefetch(gp + 32, 0, 1); __builtin_prefetch(up + 32, 0, 1); }
    __syncthreads();
    v16bf a0  = frag_ld(&aT [(wm * 32     ) * KP], lane, KP);
    v16bf a1  = frag_ld(&aT [(wm * 32 + 16) * KP], lane, KP);
    v16bf bg0 = frag_ld(&bgT[(wn * 32     ) * KP], lane, KP);
    v16bf bg1 = frag_ld(&bgT[(wn * 32 + 16) * KP], lane, KP);
    v16bf bu0 = frag_ld(&buT[(wn * 32     ) * KP], lane, KP);
    v16bf bu1 = frag_ld(&buT[(wn * 32 + 16) * KP], lane, KP);
    cg[0][0] = wmma_bf16(a0, bg0, cg[0][0]);
    cg[0][1] = wmma_bf16(a0, bg1, cg[0][1]);
    cg[1][0] = wmma_bf16(a1, bg0, cg[1][0]);
    cg[1][1] = wmma_bf16(a1, bg1, cg[1][1]);
    cu[0][0] = wmma_bf16(a0, bu0, cu[0][0]);
    cu[0][1] = wmma_bf16(a0, bu1, cu[0][1]);
    cu[1][0] = wmma_bf16(a1, bu0, cu[1][0]);
    cu[1][1] = wmma_bf16(a1, bu1, cu[1][1]);
    __syncthreads();
  }
  const int rbase = (lane >> 4) << 3, coln = lane & 15;
#pragma unroll
  for (int mi = 0; mi < 2; ++mi)
#pragma unroll
    for (int ni = 0; ni < 2; ++ni)
#pragma unroll
      for (int r = 0; r < 8; ++r) {
        const int row = m0 + wm * 32 + mi * 16 + rbase + r;
        const int col = n0 + wn * 32 + ni * 16 + coln;
        hs[(size_t)row * SIE + col] = (__bf16)silu_mul(cg[mi][ni][r], cu[mi][ni][r]);
      }
}

// ------------------------------------------------------------------
// 3) shared experts: down proj -> out[T,H] (direct store; initializes out)
//    A tile (bf16) pulled by async global->LDS copies
// ------------------------------------------------------------------
__global__ __launch_bounds__(256)
void k_shared_down(const __bf16* __restrict__ hs, const float* __restrict__ wd,
                   float* __restrict__ out) {
  const int m0 = blockIdx.x * 128, n0 = blockIdx.y * 64;
  __shared__ __attribute__((aligned(16))) __bf16 aT[128 * KPA];
  __shared__ __bf16 bT[64 * KP];
  const int tid = threadIdx.x, lane = tid & 31, wv = tid >> 5;
  const int wm = wv & 3, wn = wv >> 2;
  const int car = tid >> 1, cks = (tid & 1) * 16;       // async A: 128 rows x 32 halves
  const __bf16* hrow = hs + (size_t)(m0 + car) * SIE + cks;
  __bf16* arowp = &aT[car * KPA + cks];
  const int bn = tid >> 2, bkb = (tid & 3) * 8;
  v8f c[2][2];
#pragma unroll
  for (int i = 0; i < 2; ++i)
#pragma unroll
    for (int j = 0; j < 2; ++j) c[i][j] = (v8f){0,0,0,0,0,0,0,0};

  for (int k0 = 0; k0 < SIE; k0 += 32) {
    async_copy_b128(hrow + k0,     arowp);
    async_copy_b128(hrow + k0 + 8, arowp + 8);
    const float* wp = wd + (size_t)(n0 + bn) * SIE + k0 + bkb;
#pragma unroll
    for (int j = 0; j < 8; ++j)
      bT[bn * KP + bkb + j] = (__bf16)wp[j];
    if (k0 + 32 < SIE) __builtin_prefetch(wp + 32, 0, 1);
    wait_async0();
    __syncthreads();
    v16bf a0 = frag_ld(&aT[(wm * 32     ) * KPA], lane, KPA);
    v16bf a1 = frag_ld(&aT[(wm * 32 + 16) * KPA], lane, KPA);
    v16bf b0 = frag_ld(&bT[(wn * 32     ) * KP], lane, KP);
    v16bf b1 = frag_ld(&bT[(wn * 32 + 16) * KP], lane, KP);
    c[0][0] = wmma_bf16(a0, b0, c[0][0]);
    c[0][1] = wmma_bf16(a0, b1, c[0][1]);
    c[1][0] = wmma_bf16(a1, b0, c[1][0]);
    c[1][1] = wmma_bf16(a1, b1, c[1][1]);
    __syncthreads();
  }
  const int rbase = (lane >> 4) << 3, coln = lane & 15;
#pragma unroll
  for (int mi = 0; mi < 2; ++mi)
#pragma unroll
    for (int ni = 0; ni < 2; ++ni)
#pragma unroll
      for (int r = 0; r < 8; ++r) {
        const int row = m0 + wm * 32 + mi * 16 + rbase + r;
        const int col = n0 + wn * 32 + ni * 16 + coln;
        out[(size_t)row * HD + col] = c[mi][ni][r];
      }
}

// ------------------------------------------------------------------
// 4) expert gate+up fused SwiGLU -> hid[E,CAP,ID] bf16 (gathered rows)
// ------------------------------------------------------------------
__global__ __launch_bounds__(256)
void k_expert_gateup(const float* __restrict__ x, const float* __restrict__ gup,
                     const int* __restrict__ counts, const int* __restrict__ buf_tok,
                     __bf16* __restrict__ hid) {
  const int e = blockIdx.x, mt = blockIdx.y, nt = blockIdx.z;
  int cnt = counts[e]; if (cnt > CAP) cnt = CAP;
  const int m0 = mt * 128;
  if (m0 >= cnt) return;
  const int n0 = nt * 64;
  __shared__ __bf16 aT[128 * KP];
  __shared__ __bf16 bgT[64 * KP];
  __shared__ __bf16 buT[64 * KP];
  const int tid = threadIdx.x, lane = tid & 31, wv = tid >> 5;
  const int wm = wv & 3, wn = wv >> 2;
  const int arow = tid >> 1, aks = (tid & 1) * 16;
  const int slot = m0 + arow;
  const int tok = (slot < cnt) ? buf_tok[e * CAP + slot] : 0;
  const float* xrow = x + (size_t)tok * HD;
  const int bk = tid >> 3, bnb = (tid & 7) * 8;
  const float* Wg = gup + (size_t)e * HD * TWOI;
  v8f cg[2][2], cu[2][2];
#pragma unroll
  for (int i = 0; i < 2; ++i)
#pragma unroll
    for (int j = 0; j < 2; ++j) { cg[i][j] = (v8f){0,0,0,0,0,0,0,0}; cu[i][j] = cg[i][j]; }

  for (int k0 = 0; k0 < HD; k0 += 32) {
#pragma unroll
    for (int j = 0; j < 16; ++j)
      aT[arow * KP + aks + j] = (__bf16)xrow[k0 + aks + j];
    const float* wgp = Wg + (size_t)(k0 + bk) * TWOI + n0 + bnb;
    const float* wup = wgp + ID;
#pragma unroll
    for (int j = 0; j < 8; ++j) {
      bgT[(bnb + j) * KP + bk] = (__bf16)wgp[j];
      buT[(bnb + j) * KP + bk] = (__bf16)wup[j];
    }
    if (k0 + 32 < HD) {
      __builtin_prefetch(wgp + (size_t)32 * TWOI, 0, 1);
      __builtin_prefetch(wup + (size_t)32 * TWOI, 0, 1);
    }
    __syncthreads();
    v16bf a0  = frag_ld(&aT [(wm * 32     ) * KP], lane, KP);
    v16bf a1  = frag_ld(&aT [(wm * 32 + 16) * KP], lane, KP);
    v16bf bg0 = frag_ld(&bgT[(wn * 32     ) * KP], lane, KP);
    v16bf bg1 = frag_ld(&bgT[(wn * 32 + 16) * KP], lane, KP);
    v16bf bu0 = frag_ld(&buT[(wn * 32     ) * KP], lane, KP);
    v16bf bu1 = frag_ld(&buT[(wn * 32 + 16) * KP], lane, KP);
    cg[0][0] = wmma_bf16(a0, bg0, cg[0][0]);
    cg[0][1] = wmma_bf16(a0, bg1, cg[0][1]);
    cg[1][0] = wmma_bf16(a1, bg0, cg[1][0]);
    cg[1][1] = wmma_bf16(a1, bg1, cg[1][1]);
    cu[0][0] = wmma_bf16(a0, bu0, cu[0][0]);
    cu[0][1] = wmma_bf16(a0, bu1, cu[0][1]);
    cu[1][0] = wmma_bf16(a1, bu0, cu[1][0]);
    cu[1][1] = wmma_bf16(a1, bu1, cu[1][1]);
    __syncthreads();
  }
  const int rbase = (lane >> 4) << 3, coln = lane & 15;
#pragma unroll
  for (int mi = 0; mi < 2; ++mi)
#pragma unroll
    for (int ni = 0; ni < 2; ++ni)
#pragma unroll
      for (int r = 0; r < 8; ++r) {
        const int row = m0 + wm * 32 + mi * 16 + rbase + r;
        const int col = n0 + wn * 32 + ni * 16 + coln;
        hid[((size_t)e * CAP + row) * ID + col] =
            (__bf16)silu_mul(cg[mi][ni][r], cu[mi][ni][r]);
      }
}

// ------------------------------------------------------------------
// 5) expert down proj, scale by router weight, scatter-add into out
//    A tile (bf16) pulled by async global->LDS copies
// ------------------------------------------------------------------
__global__ __launch_bounds__(256)
void k_expert_down(const __bf16* __restrict__ hid, const float* __restrict__ dwn,
                   const int* __restrict__ counts, const int* __restrict__ buf_tok,
                   const float* __restrict__ buf_w, float* __restrict__ out) {
  const int e = blockIdx.x, mt = blockIdx.y, nt = blockIdx.z;
  int cnt = counts[e]; if (cnt > CAP) cnt = CAP;
  const int m0 = mt * 128;
  if (m0 >= cnt) return;
  const int n0 = nt * 64;
  __shared__ __attribute__((aligned(16))) __bf16 aT[128 * KPA];
  __shared__ __bf16 bT[64 * KP];
  const int tid = threadIdx.x, lane = tid & 31, wv = tid >> 5;
  const int wm = wv & 3, wn = wv >> 2;
  const int car = tid >> 1, cks = (tid & 1) * 16;
  const __bf16* hrow = hid + ((size_t)e * CAP + m0 + car) * ID + cks;
  __bf16* arowp = &aT[car * KPA + cks];
  const int bk = tid >> 3, bnb = (tid & 7) * 8;
  const float* W = dwn + (size_t)e * ID * HD;
  v8f c[2][2];
#pragma unroll
  for (int i = 0; i < 2; ++i)
#pragma unroll
    for (int j = 0; j < 2; ++j) c[i][j] = (v8f){0,0,0,0,0,0,0,0};

  for (int k0 = 0; k0 < ID; k0 += 32) {
    async_copy_b128(hrow + k0,     arowp);
    async_copy_b128(hrow + k0 + 8, arowp + 8);
    const float* wp = W + (size_t)(k0 + bk) * HD + n0 + bnb;
#pragma unroll
    for (int j = 0; j < 8; ++j)
      bT[(bnb + j) * KP + bk] = (__bf16)wp[j];
    if (k0 + 32 < ID) __builtin_prefetch(wp + (size_t)32 * HD, 0, 1);
    wait_async0();
    __syncthreads();
    v16bf a0 = frag_ld(&aT[(wm * 32     ) * KPA], lane, KPA);
    v16bf a1 = frag_ld(&aT[(wm * 32 + 16) * KPA], lane, KPA);
    v16bf b0 = frag_ld(&bT[(wn * 32     ) * KP], lane, KP);
    v16bf b1 = frag_ld(&bT[(wn * 32 + 16) * KP], lane, KP);
    c[0][0] = wmma_bf16(a0, b0, c[0][0]);
    c[0][1] = wmma_bf16(a0, b1, c[0][1]);
    c[1][0] = wmma_bf16(a1, b0, c[1][0]);
    c[1][1] = wmma_bf16(a1, b1, c[1][1]);
    __syncthreads();
  }
  const int rbase = (lane >> 4) << 3, coln = lane & 15;
#pragma unroll
  for (int mi = 0; mi < 2; ++mi)
#pragma unroll
    for (int ni = 0; ni < 2; ++ni)
#pragma unroll
      for (int r = 0; r < 8; ++r) {
        const int row = m0 + wm * 32 + mi * 16 + rbase + r;
        if (row < cnt) {
          const int   tok = buf_tok[e * CAP + row];
          const float wgt = buf_w [e * CAP + row];
          const int   col = n0 + wn * 32 + ni * 16 + coln;
          atomicAdd(&out[(size_t)tok * HD + col], c[mi][ni][r] * wgt);
        }
      }
}

// ------------------------------------------------------------------
// host launcher
// ------------------------------------------------------------------
extern "C" void kernel_launch(void* const* d_in, const int* in_sizes, int n_in,
                              void* d_out, int out_size, void* d_ws, size_t ws_size,
                              hipStream_t stream) {
  (void)in_sizes; (void)n_in; (void)out_size; (void)ws_size;
  const float* x    = (const float*)d_in[0];   // [T, H]
  const float* gw   = (const float*)d_in[1];   // [E, H]
  const float* gup  = (const float*)d_in[2];   // [E, H, 2I]
  const float* dwn  = (const float*)d_in[3];   // [E, I, H]
  const float* swg  = (const float*)d_in[4];   // [SIE, H]
  const float* swu  = (const float*)d_in[5];   // [SIE, H]
  const float* swd  = (const float*)d_in[6];   // [H, SIE]
  float* out = (float*)d_out;                  // [T, H]

  char* ws = (char*)d_ws;
  size_t off = 0;
  int*    counts  = (int*)(ws + off);    off += 256;
  int*    buf_tok = (int*)(ws + off);    off += (size_t)NE * CAP * 4;
  float*  buf_w   = (float*)(ws + off);  off += (size_t)NE * CAP * 4;
  __bf16* hid     = (__bf16*)(ws + off); off += (size_t)NE * CAP * ID * 2;
  __bf16* hs      = (__bf16*)(ws + off); off += (size_t)NTOK * SIE * 2;

  k_init<<<1, 64, 0, stream>>>(counts);
  k_router<<<NTOK, 256, 0, stream>>>(x, gw, counts, buf_tok, buf_w);
  k_shared_gateup<<<dim3(NTOK / 128, SIE / 64), 256, 0, stream>>>(x, swg, swu, hs);
  k_shared_down  <<<dim3(NTOK / 128, HD  / 64), 256, 0, stream>>>(hs, swd, out);
  k_expert_gateup<<<dim3(NE, CAP / 128, ID / 64), 256, 0, stream>>>(x, gup, counts,
                                                                    buf_tok, hid);
  k_expert_down  <<<dim3(NE, CAP / 128, HD / 64), 256, 0, stream>>>(hid, dwn, counts,
                                                                    buf_tok, buf_w, out);
}
